// GFN_60816736912089
// MI455X (gfx1250) — compile-verified, matrix operand order checked by reading, not run on hardware
//
#include <hip/hip_runtime.h>
#include <hip/hip_fp16.h>

typedef __attribute__((ext_vector_type(16))) _Float16     v16h;
typedef __attribute__((ext_vector_type(8)))  float        v8f;
typedef __attribute__((ext_vector_type(4)))  float        f32x4;
typedef __attribute__((ext_vector_type(4)))  unsigned int u32x4;
typedef __attribute__((ext_vector_type(8)))  int          i32x8;
typedef __attribute__((ext_vector_type(4)))  int          i32x4;

constexpr int   kDim   = 256;
constexpr int   kRank  = 32;
constexpr int   kDepth = 4;
constexpr int   kS     = 2048;
constexpr int   kB     = 4;
constexpr int   kVocab = 32000;
constexpr float kDt    = 0.1f;
constexpr float kEps   = 1e-5f;

#if defined(__has_builtin)
#if __has_builtin(__builtin_amdgcn_tensor_load_to_lds) && \
    __has_builtin(__builtin_amdgcn_s_wait_tensorcnt)
#define USE_TDM 1
#endif
#endif
#ifndef USE_TDM
#define USE_TDM 0
#endif

// ---------------------------------------------------------------------------
// Kernel 0: Wro f32 -> f16 (one-time; Wro_f16 = 16.4 MB stays L2-resident)
// ---------------------------------------------------------------------------
__global__ void cvt_f16_kernel(const float* __restrict__ src,
                               _Float16* __restrict__ dst, int n) {
  int i = blockIdx.x * blockDim.x + threadIdx.x;
  if (i < n) dst[i] = (_Float16)src[i];
}

// ---------------------------------------------------------------------------
// Kernel 1: sequential geodesic scan. One workgroup per batch chain (B=4).
// Thread d owns x[d], v[d] in registers; LDS mirrors state for rank-space
// reductions. Weights (393 KB) stream from L2 (trivial BW). Emits h as f16.
// ---------------------------------------------------------------------------
__global__ __launch_bounds__(256)
void scan_kernel(const int*   __restrict__ ids,
                 const float* __restrict__ emb,
                 const float* __restrict__ Win,
                 const float* __restrict__ Wx,
                 const float* __restrict__ Wout,
                 const float* __restrict__ ln_g,
                 const float* __restrict__ ln_b,
                 const float* __restrict__ x0,
                 const float* __restrict__ v0,
                 _Float16*    __restrict__ Hh,
                 float*       __restrict__ out_x,
                 float*       __restrict__ out_v) {
  const int b   = blockIdx.x;
  const int tid = threadIdx.x;

  __shared__ float Xs[kDim];
  __shared__ float Vs[kDim];
  __shared__ float Ws[kRank];
  __shared__ float Red[16];

  float xr = x0[tid];
  float vr = v0[tid];
  const float gln = ln_g[tid];
  const float bln = ln_b[tid];

  const int r     = tid >> 3;   // rank row 0..31
  const int c     = tid & 7;    // partial chunk 0..7
  const int dbase = c * 32;

  Xs[tid] = xr;
  Vs[tid] = vr;
  __syncthreads();

  for (int t = 0; t < kS; ++t) {
    const int tok = ids[b * kS + t];
    const float f = emb[(size_t)tok * kDim + tid];
    if (t + 1 < kS) {
      const int ntok = ids[b * kS + t + 1];
      __builtin_prefetch(&emb[(size_t)ntok * kDim + tid], 0, 3); // global_prefetch
    }

    for (int l = 0; l < kDepth; ++l) {
      const float* winl  = Win  + (size_t)l * kRank * kDim;
      const float* wxl   = Wx   + (size_t)l * kRank * kDim;
      const float* wo    = Wout + (size_t)l * kDim * kRank + tid * kRank;

      // ---- acc #1 on (Xs, Vs) ----
      float a1;
      {
        const float* wr = winl + r * kDim + dbase;
        const float* gr = wxl  + r * kDim + dbase;
        float pu = 0.f, pg = 0.f;
#pragma unroll
        for (int i = 0; i < 32; ++i) {
          pu += wr[i] * Vs[dbase + i];
          pg += gr[i] * Xs[dbase + i];
        }
        pu += __shfl_xor(pu, 1, 8);  pg += __shfl_xor(pg, 1, 8);
        pu += __shfl_xor(pu, 2, 8);  pg += __shfl_xor(pg, 2, 8);
        pu += __shfl_xor(pu, 4, 8);  pg += __shfl_xor(pg, 4, 8);
        if (c == 0) { float gg = tanhf(pg); Ws[r] = gg * pu * pu; }
        __syncthreads();
        float s = 0.f;
#pragma unroll
        for (int rr = 0; rr < kRank; ++rr) s += wo[rr] * Ws[rr];
        a1 = f - s;
      }

      const float xe = xr + kDt * vr;
      const float ve = vr + kDt * a1;
      Xs[tid] = xe;
      Vs[tid] = ve;
      __syncthreads();

      // ---- acc #2 on (xe, ve) ----
      float a2;
      {
        const float* wr = winl + r * kDim + dbase;
        const float* gr = wxl  + r * kDim + dbase;
        float pu = 0.f, pg = 0.f;
#pragma unroll
        for (int i = 0; i < 32; ++i) {
          pu += wr[i] * Vs[dbase + i];
          pg += gr[i] * Xs[dbase + i];
        }
        pu += __shfl_xor(pu, 1, 8);  pg += __shfl_xor(pg, 1, 8);
        pu += __shfl_xor(pu, 2, 8);  pg += __shfl_xor(pg, 2, 8);
        pu += __shfl_xor(pu, 4, 8);  pg += __shfl_xor(pg, 4, 8);
        if (c == 0) { float gg = tanhf(pg); Ws[r] = gg * pu * pu; }
        __syncthreads();
        float s = 0.f;
#pragma unroll
        for (int rr = 0; rr < kRank; ++rr) s += wo[rr] * Ws[rr];
        a2 = f - s;
      }

      xr = xr + 0.5f * kDt * (vr + ve);
      vr = vr + 0.5f * kDt * (a1 + a2);
      Xs[tid] = xr;
      Vs[tid] = vr;
      __syncthreads();
    }

    // ---- layernorm over 256 dims ----
    float s1 = xr, s2 = xr * xr;
#pragma unroll
    for (int off = 16; off >= 1; off >>= 1) {
      s1 += __shfl_xor(s1, off, 32);
      s2 += __shfl_xor(s2, off, 32);
    }
    const int wv = tid >> 5;
    if ((tid & 31) == 0) { Red[wv] = s1; Red[8 + wv] = s2; }
    __syncthreads();
    float m1 = 0.f, m2 = 0.f;
#pragma unroll
    for (int w = 0; w < 8; ++w) { m1 += Red[w]; m2 += Red[8 + w]; }
    m1 *= (1.f / kDim);
    m2 = m2 * (1.f / kDim) - m1 * m1;
    const float h = (xr - m1) * rsqrtf(m2 + kEps) * gln + bln;
    Hh[((size_t)b * kS + t) * kDim + tid] = (_Float16)h;
    __syncthreads();  // Red reuse next step
  }

  out_x[b * kDim + tid] = xr;
  out_v[b * kDim + tid] = vr;
}

// ---------------------------------------------------------------------------
// Kernel 2: readout GEMM  logits[8192,32000] = H[8192,256] * Wro^T + bro
// v_wmma_f32_16x16x32_f16; block tile 64x128, 8 waves, 32x32 per wave.
// K staged in 64-wide chunks, double-buffered via the Tensor Data Mover
// (tensor_load_to_lds + s_wait_tensorcnt). TDM padding (interval=32 DWORDs,
// amount=4 DWORDs) produces the bank-skewed LDS row stride of 72 halfs.
// Epilogue restages C through LDS and streams full-line nontemporal b128.
// ---------------------------------------------------------------------------
constexpr int BM  = 64;
constexpr int BN  = 128;
constexpr int KC  = 64;
constexpr int LDT = 72;                         // halfs per LDS row (144 B)
constexpr int ABYTES   = BM * LDT * 2;          // 9216
constexpr int BBYTES   = BN * LDT * 2;          // 18432
constexpr int BUFBYTES = ABYTES + BBYTES;       // 27648
constexpr int CLDF     = BN + 4;                // C staging stride (floats)
constexpr int SMEM_SZ  = 2 * BUFBYTES;          // 55296 >= BM*CLDF*4 (33792)

#if USE_TDM
__device__ __forceinline__ void tdm_issue_tile(unsigned int lds_addr,
                                               const _Float16* gptr,
                                               int tensor_rows, int tile_rows) {
  const unsigned long long ga = (unsigned long long)(uintptr_t)gptr;
  u32x4 g0 = { 1u,                                   // count=1 (valid user D#)
               lds_addr,                             // LDS byte address
               (unsigned int)ga,                     // global_addr[31:0]
               ((unsigned int)(ga >> 32) & 0x01FFFFFFu) | (2u << 30) }; // type=2
  // group1: data_size=2B (code 1), pad_enable, pad_interval=32 DW (code 4),
  // pad_amount=4 DW (code 3); tensor_dim0=256, tile_dim0=64, stride0=256.
  i32x8 g1 = { (int)((1u << 16) | (1u << 20) | (4u << 22) | (3u << 25)),
               (int)(256u << 16),                          // tensor_dim0[15:0]
               (int)((unsigned int)tensor_rows << 16),     // tensor_dim1[15:0]
               (int)(64u << 16),                           // tile_dim0
               tile_rows,                                  // tile_dim1
               256,                                        // dim0_stride[31:0]
               0, 0 };
  i32x4 gz = { 0, 0, 0, 0 };
#if __clang_major__ >= 23
  i32x8 gz8 = { 0, 0, 0, 0, 0, 0, 0, 0 };
  __builtin_amdgcn_tensor_load_to_lds(g0, g1, gz, gz, gz8, 0);
#else
  __builtin_amdgcn_tensor_load_to_lds(g0, g1, gz, gz, 0);
#endif
}
#endif

__global__ __launch_bounds__(256)
void readout_gemm_kernel(const _Float16* __restrict__ A,    // H  [8192, 256]
                         const _Float16* __restrict__ Bm,   // Wro[32000, 256]
                         const float*    __restrict__ bro,
                         float*          __restrict__ C) {  // [8192, 32000]
  __shared__ __align__(16) char smem[SMEM_SZ];

  const int tid  = threadIdx.x;
  const int lane = tid & 31;
  const int wid  = tid >> 5;
  const int wm   = wid >> 2;      // 0..1
  const int wn   = wid & 3;       // 0..3
  const int m0   = blockIdx.x * BM;
  const int n0   = blockIdx.y * BN;
  const int l15  = lane & 15;
  const int hsel = lane >> 4;     // half-wave select per ISA fragment layout

  v8f acc[2][2] = {};

#if USE_TDM
  const unsigned int smb = (unsigned int)(uintptr_t)&smem[0];
  if (wid == 0) {   // one wave drives the DMA for the whole block
    tdm_issue_tile(smb,          A  + (size_t)m0 * kDim, kB * kS, BM);
    tdm_issue_tile(smb + ABYTES, Bm + (size_t)n0 * kDim, kVocab, BN);
  }
#endif

  for (int ci = 0; ci < kDim / KC; ++ci) {
    const int cur = ci & 1;
#if USE_TDM
    if (wid == 0) {
      if (ci + 1 < kDim / KC) {
        const int nxt = (ci + 1) & 1;
        tdm_issue_tile(smb + nxt * BUFBYTES,
                       A  + (size_t)m0 * kDim + (ci + 1) * KC, kB * kS, BM);
        tdm_issue_tile(smb + nxt * BUFBYTES + ABYTES,
                       Bm + (size_t)n0 * kDim + (ci + 1) * KC, kVocab, BN);
        __builtin_amdgcn_s_wait_tensorcnt(2);   // chunk ci complete (in-order)
      } else {
        __builtin_amdgcn_s_wait_tensorcnt(0);
      }
    }
    __syncthreads();
#else
    {
      _Float16* Ash = (_Float16*)(smem + cur * BUFBYTES);
      _Float16* Bsh = (_Float16*)(smem + cur * BUFBYTES + ABYTES);
      const int kc = ci * KC;
      for (int idx = tid; idx < (BM * KC) / 8; idx += 256) {
        const int row = idx >> 3, col = (idx & 7) * 8;
        *(uint4*)(&Ash[row * LDT + col]) =
            *(const uint4*)(&A[(size_t)(m0 + row) * kDim + kc + col]);
      }
      for (int idx = tid; idx < (BN * KC) / 8; idx += 256) {
        const int row = idx >> 3, col = (idx & 7) * 8;
        *(uint4*)(&Bsh[row * LDT + col]) =
            *(const uint4*)(&Bm[(size_t)(n0 + row) * kDim + kc + col]);
      }
      __syncthreads();
    }
#endif

    const _Float16* Acur = (const _Float16*)(smem + cur * BUFBYTES);
    const _Float16* Bcur = (const _Float16*)(smem + cur * BUFBYTES + ABYTES);

#pragma unroll
    for (int kk = 0; kk < KC; kk += 32) {
      const int kb = kk + hsel * 8;   // lanes 0-15: K{0..7,16..23}; 16-31: +8
      v16h af[2], bf[2];
#pragma unroll
      for (int i = 0; i < 2; ++i) {
        const int mr = wm * 32 + i * 16 + l15;
        union { v16h v; uint4 q[2]; } ua;
        ua.q[0] = *(const uint4*)(&Acur[mr * LDT + kb]);
        ua.q[1] = *(const uint4*)(&Acur[mr * LDT + kb + 16]);
        af[i] = ua.v;
        const int nr = wn * 32 + i * 16 + l15;
        union { v16h v; uint4 q[2]; } ub;
        ub.q[0] = *(const uint4*)(&Bcur[nr * LDT + kb]);
        ub.q[1] = *(const uint4*)(&Bcur[nr * LDT + kb + 16]);
        bf[i] = ub.v;
      }
#pragma unroll
      for (int i = 0; i < 2; ++i)
#pragma unroll
        for (int j = 0; j < 2; ++j)
          acc[i][j] = __builtin_amdgcn_wmma_f32_16x16x32_f16(
              false, af[i], false, bf[j], (short)0, acc[i][j], false, false);
    }
    __syncthreads();   // buffer free before it is re-filled
  }

  // ---- epilogue: restage C tile in LDS, stream out nontemporal b128 ----
  float* Cs = (float*)smem;   // 64 x CLDF floats = 33792 B (aliases buffers)
#pragma unroll
  for (int i = 0; i < 2; ++i) {
    const int mr = wm * 32 + i * 16 + hsel * 8;   // C layout: VGPR p -> M=p / M=8+p
#pragma unroll
    for (int j = 0; j < 2; ++j) {
      const int nc = wn * 32 + j * 16 + l15;
#pragma unroll
      for (int p = 0; p < 8; ++p)
        Cs[(mr + p) * CLDF + nc] = acc[i][j][p];
    }
  }
  __syncthreads();

  for (int idx = tid; idx < BM * (BN / 4); idx += 256) {   // 8 iterations
    const int row  = idx >> 5;          // 0..63
    const int col4 = (idx & 31) * 4;    // 0..124
    f32x4 val = *(const f32x4*)&Cs[row * CLDF + col4];
    f32x4 bb  = *(const f32x4*)&bro[n0 + col4];
    f32x4 o   = val + bb;
    __builtin_nontemporal_store(
        o, (f32x4*)&C[(size_t)(m0 + row) * kVocab + n0 + col4]);
  }
}

// ---------------------------------------------------------------------------
extern "C" void kernel_launch(void* const* d_in, const int* in_sizes, int n_in,
                              void* d_out, int out_size, void* d_ws, size_t ws_size,
                              hipStream_t stream) {
  const int*   ids  = (const int*)  d_in[0];
  const float* emb  = (const float*)d_in[1];
  const float* Win  = (const float*)d_in[2];
  const float* Wx   = (const float*)d_in[3];
  const float* Wout = (const float*)d_in[4];
  const float* ln_g = (const float*)d_in[5];
  const float* ln_b = (const float*)d_in[6];
  const float* Wro  = (const float*)d_in[7];
  const float* bro  = (const float*)d_in[8];
  const float* x0   = (const float*)d_in[9];
  const float* v0   = (const float*)d_in[10];

  float* logits = (float*)d_out;                       // [B, S, VOCAB]
  float* out_x  = logits + (size_t)kB * kS * kVocab;   // [B, DIM]
  float* out_v  = out_x + (size_t)kB * kDim;           // [B, DIM]

  // workspace: H_f16 (4 MB) then Wro_f16 (16.4 MB)
  _Float16* Hh   = (_Float16*)d_ws;
  _Float16* Wroh = (_Float16*)((char*)d_ws +
                               (size_t)kB * kS * kDim * sizeof(_Float16));

  const int nwro = kVocab * kDim;
  cvt_f16_kernel<<<(nwro + 255) / 256, 256, 0, stream>>>(Wro, Wroh, nwro);

  scan_kernel<<<kB, 256, 0, stream>>>(ids, emb, Win, Wx, Wout, ln_g, ln_b,
                                      x0, v0, Hh, out_x, out_v);

  dim3 grid((kB * kS) / BM, kVocab / BN);   // 128 x 250 = 32000 blocks
  readout_gemm_kernel<<<grid, 256, 0, stream>>>(Hh, Wroh, bro, logits);
}
